// HeteroLinkPredictionModel_8083128451632
// MI455X (gfx1250) — matrix operand order repaired and checked by reading, not compiled, not run on hardware
//
#include <hip/hip_runtime.h>
#include <hip/hip_bf16.h>
#include <math.h>

// ---------------------------------------------------------------------------
// HeteroLinkPredictionModel inference for MI455X (gfx1250, wave32, WMMA).
//
// Per (layer, edge-type), one dense bf16 WMMA GEMM computes Y = h @ [W2|W3|W4]
// over all nodes; a per-edge VALU kernel applies the sigmoid gate and scatters
// messages with global f32 atomics. Skip / JK / MLP layers are WMMA GEMMs with
// fused bias+ReLU epilogues.  GEMM uses a 32x64 output tile per wave
// (2x4 accumulator tiles, 8 v_wmma per K-step, 12 b128 fragment loads)
// so the loop is WMMA-throughput-bound rather than cache-port-bound.
//
// d_in order (setup_inputs dict order, depth-first):
//  0 target_edge_index(2,ET) i32   1 x(N,128) f32      2 embed_edge_index(2,E) i32
//  3 edge_type(E) i32              4 pitch(ET,1) f32   5 onset(ET,2) f32
//  6..10  conv0: Wskip(128,256) W2(7,128,256) W3 W4 b(256)
//  11..15 conv1: Wskip(256,256) W2(7,256,256) W3 W4 b(256)
//  16..20 conv2: same shapes as conv1
//  21 ln_g(256) 22 ln_b(256) 23 jk_W(768,256) 24 jk_b(256)
//  25 p1_W(515,256) 26 p1_b(256) 27 p2_W(256,128) 28 p2_b(128)
//  29 p3_W(128,1) 30 p3_b(1)
// d_out: (ET,1) f32.  Requires ws_size >= ~785 MB.
// ---------------------------------------------------------------------------

#define N_NODES 100000
#define N_EDGES 300000
#define N_TEDGES 200000
#define HDIM 256
#define NTYPES 7

typedef __attribute__((ext_vector_type(16))) __bf16 v16bf;
typedef __attribute__((ext_vector_type(8)))  float  v8f;

union FragBf { v16bf v; float4 f[2]; };

// ---------------------------------------------------------------------------
// bf16 WMMA GEMM:  C[M,Nc] = A[M,K](bf16,row-major) * BT[Nc,K](bf16)^T
// + optional bias[Nc] + optional ReLU.  Writes fp32 C and/or bf16 Cb.
// Requires M%32==0, Nc%64==0, K%32==0 (caller pads).  One 32x64 output tile
// per wave (2x4 WMMA tiles); 4 waves (128 threads) per block.
// ---------------------------------------------------------------------------
__global__ void __launch_bounds__(128)
k_wmma_gemm(const __bf16* __restrict__ A, const __bf16* __restrict__ BT,
            const float* __restrict__ bias, float* __restrict__ C,
            __bf16* __restrict__ Cb, int M, int Nc, int K, int relu)
{
    const int tilesN = Nc >> 6;
    const int tiles  = (M >> 5) * tilesN;
    const int tile   = blockIdx.x * 4 + (threadIdx.x >> 5);
    if (tile >= tiles) return;                  // wave-uniform: EXEC stays all-1s
    const int mt = tile / tilesN, nt = tile - mt * tilesN;
    const int m0 = mt << 5, n0 = nt << 6;
    const int lane = threadIdx.x & 31;
    const int r  = lane & 15;                   // row (A) / column (B) within tile
    const int kh = lane >> 4;                   // K half-select per ISA layout

    // A frag: lanes 0-15 carry K {0..7,16..23}, lanes 16-31 K {8..15,24..31}
    const __bf16* pA[2];
    pA[0] = A + (size_t)(m0 + r) * K + (kh << 3);
    pA[1] = pA[0] + (size_t)16 * K;
    // B frag: lanes 0-15 carry K {0..7,16..23}, lanes 16-31 K {8..15,24..31}; col = n0+16j+r
    const __bf16* pB[4];
    pB[0] = BT + (size_t)(n0 + r) * K + (kh << 3);
    pB[1] = pB[0] + (size_t)16 * K;
    pB[2] = pB[0] + (size_t)32 * K;
    pB[3] = pB[0] + (size_t)48 * K;

    v8f acc[2][4];
#pragma unroll
    for (int i = 0; i < 2; ++i)
#pragma unroll
        for (int j = 0; j < 4; ++j)
            acc[i][j] = (v8f){0.f, 0.f, 0.f, 0.f, 0.f, 0.f, 0.f, 0.f};

    for (int k = 0; k < K; k += 32) {
        FragBf a[2], b[4];
#pragma unroll
        for (int i = 0; i < 2; ++i) {
            a[i].f[0] = *(const float4*)(pA[i] + k);
            a[i].f[1] = *(const float4*)(pA[i] + k + 16);
        }
#pragma unroll
        for (int j = 0; j < 4; ++j) {
            b[j].f[0] = *(const float4*)(pB[j] + k);
            b[j].f[1] = *(const float4*)(pB[j] + k + 16);
        }
#pragma unroll
        for (int i = 0; i < 2; ++i)
#pragma unroll
            for (int j = 0; j < 4; ++j)
                acc[i][j] = __builtin_amdgcn_wmma_f32_16x16x32_bf16(
                                false, a[i].v, false, b[j].v, (short)0,
                                acc[i][j], false, false);
    }

    const int row0 = m0 + (kh << 3);            // acc VGPR e -> row base + e (+8 hi half)
#pragma unroll
    for (int i = 0; i < 2; ++i) {
        const int rowBase = row0 + i * 16;
#pragma unroll
        for (int j = 0; j < 4; ++j) {
            const int col = n0 + j * 16 + r;
            const float bv = bias ? bias[col] : 0.f;
#pragma unroll
            for (int e = 0; e < 8; ++e) {
                float v = acc[i][j][e] + bv;
                if (relu && v < 0.f) v = 0.f;
                const size_t idx = (size_t)(rowBase + e) * Nc + col;
                if (C)  C[idx]  = v;
                if (Cb) Cb[idx] = (__bf16)v;
            }
        }
    }
}

// ------------------------- small utility kernels ---------------------------
__global__ void k_zero(float* p, size_t n)
{
    size_t i = (size_t)blockIdx.x * blockDim.x + threadIdx.x;
    if (i < n) p[i] = 0.f;
}

__global__ void k_cvt_bf16(const float* __restrict__ a, __bf16* __restrict__ b, size_t n)
{
    size_t i = (size_t)blockIdx.x * blockDim.x + threadIdx.x;
    if (i < n) b[i] = (__bf16)a[i];
}

// WT[n*Kpad+k] = (k<K) ? W[k*Nc+n] : 0   (fp32 [K,Nc] -> bf16 [Nc,Kpad])
__global__ void k_transpose_bf16(const float* __restrict__ W, __bf16* __restrict__ WT,
                                 int K, int Nc, int Kpad)
{
    size_t i = (size_t)blockIdx.x * blockDim.x + threadIdx.x;
    size_t total = (size_t)Nc * Kpad;
    if (i >= total) return;
    int n = (int)(i / Kpad), k = (int)(i % Kpad);
    WT[i] = (k < K) ? (__bf16)W[(size_t)k * Nc + n] : (__bf16)0.0f;
}

// Concatenated gate/message weights: dst[t][n][k] = W{2,3,4}[t][k][n%256]
// dst is bf16 [T][768][din]; rows 0-255 <- W2, 256-511 <- W3, 512-767 <- W4.
__global__ void k_prep_cat(const float* __restrict__ W2, const float* __restrict__ W3,
                           const float* __restrict__ W4, __bf16* __restrict__ dst, int din)
{
    size_t i = (size_t)blockIdx.x * blockDim.x + threadIdx.x;
    size_t total = (size_t)NTYPES * 768 * din;
    if (i >= total) return;
    int t   = (int)(i / ((size_t)768 * din));
    int rem = (int)(i % ((size_t)768 * din));
    int n = rem / din, k = rem - n * din;
    const float* Ws = (n < 256) ? W2 : ((n < 512) ? W3 : W4);
    dst[i] = (__bf16)Ws[(size_t)t * din * HDIM + (size_t)k * HDIM + (n & 255)];
}

// Per-edge gated message + scatter-add.  Y rows are [Y2|Y3|Y4] (768 fp32).
__global__ void k_edge_msg(const int* __restrict__ eidx, const int* __restrict__ etype,
                           const float* __restrict__ Y, float* __restrict__ agg,
                           int t, int E)
{
    size_t tid = (size_t)blockIdx.x * blockDim.x + threadIdx.x;
    int e  = (int)(tid >> 6);                   // 64 chunks of 4 dims = 256
    int c4 = (int)(tid & 63) << 2;
    if (e >= E) return;
    if (etype[e] != t) return;
    const int s = eidx[e], d = eidx[E + e];
    const float4 y2 = *(const float4*)(Y + (size_t)s * 768 + c4);
    const float4 y3 = *(const float4*)(Y + (size_t)d * 768 + 256 + c4);
    const float4 y4 = *(const float4*)(Y + (size_t)s * 768 + 512 + c4);
    float* ap = agg + (size_t)d * HDIM + c4;
    atomicAdd(ap + 0, y2.x / (1.f + __expf(-(y3.x + y4.x))));
    atomicAdd(ap + 1, y2.y / (1.f + __expf(-(y3.y + y4.y))));
    atomicAdd(ap + 2, y2.z / (1.f + __expf(-(y3.z + y4.z))));
    atomicAdd(ap + 3, y2.w / (1.f + __expf(-(y3.w + y4.w))));
}

// h[row] = h[row] + agg[row]; if do_act: ReLU then LayerNorm(g,b).
// One wave32 per 256-wide row; reductions via __shfl_xor.
__global__ void k_post_ln(float* __restrict__ h, const float* __restrict__ agg,
                          const float* __restrict__ g, const float* __restrict__ bt,
                          int nrows, int do_act)
{
    const int wid  = (int)(((size_t)blockIdx.x * blockDim.x + threadIdx.x) >> 5);
    const int lane = threadIdx.x & 31;
    if (wid >= nrows) return;
    float* row = h + (size_t)wid * HDIM;
    const float* ar = agg + (size_t)wid * HDIM;
    float v[8];
    float s = 0.f;
#pragma unroll
    for (int i = 0; i < 8; ++i) {
        const int c = i * 32 + lane;
        float x = row[c] + ar[c];
        if (do_act && x < 0.f) x = 0.f;
        v[i] = x; s += x;
    }
    if (!do_act) {
#pragma unroll
        for (int i = 0; i < 8; ++i) row[i * 32 + lane] = v[i];
        return;
    }
#pragma unroll
    for (int off = 16; off; off >>= 1) s += __shfl_xor(s, off, 32);
    const float mu = s * (1.f / 256.f);
    float var = 0.f;
#pragma unroll
    for (int i = 0; i < 8; ++i) { float d = v[i] - mu; var += d * d; }
#pragma unroll
    for (int off = 16; off; off >>= 1) var += __shfl_xor(var, off, 32);
    const float rs = rsqrtf(var * (1.f / 256.f) + 1e-5f);
#pragma unroll
    for (int i = 0; i < 8; ++i) {
        const int c = i * 32 + lane;
        row[c] = (v[i] - mu) * rs * g[c] + bt[c];
    }
}

// JumpingKnowledge input: bf16 [N,768] concat of the three layer outputs.
__global__ void k_build_jk(const float* __restrict__ h0, const float* __restrict__ h1,
                           const float* __restrict__ h2, __bf16* __restrict__ out, int n)
{
    size_t i = (size_t)blockIdx.x * blockDim.x + threadIdx.x;
    size_t total = (size_t)n * 768;
    if (i >= total) return;
    int row = (int)(i / 768), c = (int)(i % 768);
    const float* hp = (c < 256) ? h0 : ((c < 512) ? h1 : h2);
    out[i] = (__bf16)hp[(size_t)row * HDIM + (c & 255)];
}

// Edge MLP input z: bf16 [ET,544] = [h[s] | h[d] | pitch | onset | zero-pad].
__global__ void k_build_z(const int* __restrict__ tgt, const float* __restrict__ hjk,
                          const float* __restrict__ pitch, const float* __restrict__ onset,
                          __bf16* __restrict__ z, int ET)
{
    size_t i = (size_t)blockIdx.x * blockDim.x + threadIdx.x;
    size_t total = (size_t)ET * 544;
    if (i >= total) return;
    int e = (int)(i / 544), c = (int)(i % 544);
    float v;
    if      (c < 256) v = hjk[(size_t)tgt[e] * HDIM + c];
    else if (c < 512) v = hjk[(size_t)tgt[ET + e] * HDIM + (c - 256)];
    else if (c == 512) v = pitch[e];
    else if (c == 513) v = onset[2 * e];
    else if (c == 514) v = onset[2 * e + 1];
    else               v = 0.f;                 // pad K 515 -> 544 (mult of 32)
    z[i] = (__bf16)v;
}

// Final projection to 1 dim + sigmoid (Nc=1: plain dot, no WMMA needed).
__global__ void k_p3(const __bf16* __restrict__ z2, const float* __restrict__ w,
                     const float* __restrict__ b, float* __restrict__ out, int ET)
{
    int e = blockIdx.x * blockDim.x + threadIdx.x;
    if (e >= ET) return;
    const __bf16* zr = z2 + (size_t)e * 128;
    float s = b[0];
#pragma unroll
    for (int k = 0; k < 128; ++k) s += (float)zr[k] * w[k];
    out[e] = 1.f / (1.f + __expf(-s));
}

// ---------------------------------------------------------------------------
static inline unsigned nb(size_t n, int b) { return (unsigned)((n + b - 1) / b); }

static inline void gemm(const __bf16* A, const __bf16* BT, const float* bias,
                        float* C, __bf16* Cb, int M, int Nc, int K, int relu,
                        hipStream_t s)
{
    const int tiles = (M / 32) * (Nc / 64);
    k_wmma_gemm<<<(tiles + 3) / 4, 128, 0, s>>>(A, BT, bias, C, Cb, M, Nc, K, relu);
}

extern "C" void kernel_launch(void* const* d_in, const int* in_sizes, int n_in,
                              void* d_out, int out_size, void* d_ws, size_t ws_size,
                              hipStream_t stream)
{
    (void)in_sizes; (void)n_in; (void)out_size; (void)ws_size;
    const int N = N_NODES, E = N_EDGES, ET = N_TEDGES;

    const int*   tgt    = (const int*)  d_in[0];
    const float* x      = (const float*)d_in[1];
    const int*   eidx   = (const int*)  d_in[2];
    const int*   etype  = (const int*)  d_in[3];
    const float* pitch  = (const float*)d_in[4];
    const float* onset  = (const float*)d_in[5];
    const float* Wskip[3] = {(const float*)d_in[6],  (const float*)d_in[11], (const float*)d_in[16]};
    const float* W2[3]    = {(const float*)d_in[7],  (const float*)d_in[12], (const float*)d_in[17]};
    const float* W3[3]    = {(const float*)d_in[8],  (const float*)d_in[13], (const float*)d_in[18]};
    const float* W4[3]    = {(const float*)d_in[9],  (const float*)d_in[14], (const float*)d_in[19]};
    const float* bcv[3]   = {(const float*)d_in[10], (const float*)d_in[15], (const float*)d_in[20]};
    const float* ln_g  = (const float*)d_in[21];
    const float* ln_b  = (const float*)d_in[22];
    const float* jk_W  = (const float*)d_in[23];
    const float* jk_b  = (const float*)d_in[24];
    const float* p1_W  = (const float*)d_in[25];
    const float* p1_b  = (const float*)d_in[26];
    const float* p2_W  = (const float*)d_in[27];
    const float* p2_b  = (const float*)d_in[28];
    const float* p3_W  = (const float*)d_in[29];
    const float* p3_b  = (const float*)d_in[30];

    // ---- workspace carve-up (all offsets 256B-aligned) ----
    char* ws = (char*)d_ws;
    const size_t o_w   = 0;                               // bf16 weights (~8 MB)
    const size_t o_hb  = 16777216;                        // bf16 h        (51.2 MB)
    const size_t o_hs  = o_hb + 51200000;                 // 3x fp32 hs    (307.2 MB)
    const size_t hsStride = 102400000;
    const size_t o_Y   = o_hs + 3 * hsStride;             // fp32 Y [N,768] (307.2 MB)
    const size_t o_agg = o_Y + 307200000;                 // fp32 agg      (102.4 MB)

    __bf16* wb  = (__bf16*)(ws + o_w);
    __bf16* hb  = (__bf16*)(ws + o_hb);
    float*  hs0 = (float*)(ws + o_hs);
    float*  hs1 = (float*)(ws + o_hs + hsStride);
    float*  hs2 = (float*)(ws + o_hs + 2 * hsStride);
    float*  hsL[3] = {hs0, hs1, hs2};
    float*  Y   = (float*)(ws + o_Y);
    float*  agg = (float*)(ws + o_agg);

    // ---- one-time (per launch) weight transpose/convert to bf16 ----
    const int dins[3] = {128, 256, 256};
    __bf16* skipT[3]; __bf16* catT[3];
    size_t wo = 0;
    for (int l = 0; l < 3; ++l) {
        const int din = dins[l];
        skipT[l] = wb + wo;
        k_transpose_bf16<<<nb((size_t)HDIM * din, 256), 256, 0, stream>>>(
            Wskip[l], skipT[l], din, HDIM, din);
        wo += (size_t)HDIM * din;
        catT[l] = wb + wo;
        k_prep_cat<<<nb((size_t)NTYPES * 768 * din, 256), 256, 0, stream>>>(
            W2[l], W3[l], W4[l], catT[l], din);
        wo += (size_t)NTYPES * 768 * din;
    }
    __bf16* jkT = wb + wo;
    k_transpose_bf16<<<nb((size_t)HDIM * 768, 256), 256, 0, stream>>>(jk_W, jkT, 768, HDIM, 768);
    wo += (size_t)HDIM * 768;
    __bf16* p1T = wb + wo;
    k_transpose_bf16<<<nb((size_t)HDIM * 544, 256), 256, 0, stream>>>(p1_W, p1T, 515, HDIM, 544);
    wo += (size_t)HDIM * 544;
    __bf16* p2T = wb + wo;
    k_transpose_bf16<<<nb((size_t)128 * 256, 256), 256, 0, stream>>>(p2_W, p2T, 256, 128, 256);

    // ---- 3 gated conv layers ----
    for (int l = 0; l < 3; ++l) {
        const int din = dins[l];
        const float* hin = (l == 0) ? x : hsL[l - 1];
        k_cvt_bf16<<<nb((size_t)N * din, 256), 256, 0, stream>>>(hin, hb, (size_t)N * din);
        // skip connection + bias: hs[l] = h @ Wskip + b
        gemm(hb, skipT[l], bcv[l], hsL[l], nullptr, N, HDIM, din, 0, stream);
        k_zero<<<nb((size_t)N * HDIM, 256), 256, 0, stream>>>(agg, (size_t)N * HDIM);
        for (int t = 0; t < NTYPES; ++t) {
            // Y[N,768] = h @ [W2[t]|W3[t]|W4[t]]
            gemm(hb, catT[l] + (size_t)t * 768 * din, nullptr, Y, nullptr, N, 768, din, 0, stream);
            k_edge_msg<<<nb((size_t)E * 64, 256), 256, 0, stream>>>(eidx, etype, Y, agg, t, E);
        }
        k_post_ln<<<nb((size_t)N * 32, 256), 256, 0, stream>>>(
            hsL[l], agg, ln_g, ln_b, N, (l < 2) ? 1 : 0);
    }

    // ---- JumpingKnowledge: h_jk = concat(hs) @ jk_W + jk_b ----
    __bf16* jkin = (__bf16*)(ws + o_Y);                   // reuse Y region
    k_build_jk<<<nb((size_t)N * 768, 256), 256, 0, stream>>>(hs0, hs1, hs2, jkin, N);
    float* hjk = agg;                                     // reuse agg region
    gemm(jkin, jkT, jk_b, hjk, nullptr, N, HDIM, 768, 0, stream);

    // ---- edge MLP: z -> relu(p1) -> relu(p2) -> sigmoid(p3) ----
    __bf16* z = (__bf16*)(ws + o_Y);                      // reuse Y region (jkin dead)
    k_build_z<<<nb((size_t)ET * 544, 256), 256, 0, stream>>>(tgt, hjk, pitch, onset, z, ET);
    __bf16* z1 = (__bf16*)(ws + o_hs);                    // hs region dead after JK
    gemm(z, p1T, p1_b, nullptr, z1, ET, HDIM, 544, 1, stream);
    __bf16* z2 = (__bf16*)(ws + o_hs + hsStride);
    gemm(z1, p2T, p2_b, nullptr, z2, ET, 128, HDIM, 1, stream);
    k_p3<<<nb((size_t)ET, 256), 256, 0, stream>>>(z2, p3_W, p3_b, (float*)d_out, ET);
}
